// Mamba2Block_55473797595674
// MI455X (gfx1250) — compile-verified, matrix operand order checked by reading, not run on hardware
//
#include <hip/hip_runtime.h>

#define HIDDEN 1024
#define INNER  2048
#define STATE  64
#define KCONV  4
#define BATCH  2
#define SEQ    2048

typedef __attribute__((ext_vector_type(16))) __bf16 v16bf;
typedef __attribute__((ext_vector_type(8)))  float  v8f;

__device__ __forceinline__ unsigned short f2bf(float f) {
    // round-to-nearest-even f32 -> bf16
    unsigned int u = __float_as_uint(f);
    unsigned int r = u + 0x7FFFu + ((u >> 16) & 1u);
    return (unsigned short)(r >> 16);
}

union BfFrag {
    v16bf v;
    unsigned int u[8];
};

// LDS byte-offset of a __shared__ object (AS3 pointers are 32-bit offsets
// within the workgroup's LDS allocation) — needed for async global->LDS DMA.
__device__ __forceinline__ unsigned int lds_off_u32(const void* p) {
    return (unsigned int)(unsigned long long)
        (__attribute__((address_space(3))) const void*)p;
}

// pack-4 f32 -> bf16 conversion (all sizes here are multiples of 4)
__global__ void cvt_bf16_kernel(const float* __restrict__ in,
                                unsigned short* __restrict__ out, int n) {
    int i = (blockIdx.x * blockDim.x + threadIdx.x) * 4;
    if (i >= n) return;
    float4 f = *(const float4*)(in + i);
    unsigned long long p = (unsigned long long)f2bf(f.x)
                         | ((unsigned long long)f2bf(f.y) << 16)
                         | ((unsigned long long)f2bf(f.z) << 32)
                         | ((unsigned long long)f2bf(f.w) << 48);
    *(unsigned long long*)(out + i) = p;
}

// C[M,N] = A[M,K] @ B[K,N], bf16 in (pre-converted), f32 out.
// Block tile 64 x (2*WN), 8 wave32s: wave_m 0..3 (16 rows each),
// wave_n 0..1 (WN cols each -> WN/16 WMMA accumulators per wave).
// A tile staged via async global->LDS DMA (ASYNCcnt); B tile staged
// transposed ([n][k]) through VGPRs so fragment gathers are 32-bit pairs.
// EPILOGUE: 0 = store raw; 1 = softplus(acc + bias[n]).
template <int EPILOGUE, int WN>
__global__ __launch_bounds__(256)
void gemm_bf16_wmma(const unsigned short* __restrict__ A,
                    const unsigned short* __restrict__ B,
                    const float* __restrict__ bias, float* __restrict__ C,
                    int M, int N, int K) {
    constexpr int BN  = 2 * WN;
    constexpr int NT  = WN / 16;     // WMMA n-tiles per wave
    constexpr int EPT = BN / 8;      // B elements per thread per k-row chunk

    __shared__ unsigned short As[64][32];   // [m][k]
    __shared__ unsigned short Bs[BN][32];   // [n][k] (transposed tile)

    const int tid    = threadIdx.x;
    const int lane   = tid & 31;
    const int wave   = tid >> 5;
    const int wave_m = wave & 3;
    const int wave_n = wave >> 2;
    const int half   = lane >> 4;
    const int ln     = lane & 15;

    const int bm = blockIdx.y * 64;
    const int bn = blockIdx.x * BN;

    v8f acc[NT];
#pragma unroll
    for (int t = 0; t < NT; ++t) acc[t] = {};

    // A loader: 64 rows x 32 k of bf16; 4 threads/row, 16 bytes (8 elts) each
    const int ar = tid >> 2;
    const int ak = (tid & 3) * 8;
    const unsigned int a_dst = lds_off_u32(&As[ar][ak]);
    // B loader: 32 k-rows x BN cols; 8 threads/row, EPT contiguous elts each
    const int bk  = tid >> 3;
    const int bnb = (tid & 7) * EPT;

    for (int k0 = 0; k0 < K; k0 += 32) {
        __syncthreads();   // previous iteration's LDS reads complete
        {   // async DMA of the A tile straight into LDS (no VGPR round-trip)
            const unsigned short* ap = A + (size_t)(bm + ar) * K + k0 + ak;
            asm volatile("global_load_async_to_lds_b128 %0, %1, off"
                         :: "v"(a_dst), "v"(ap) : "memory");
        }
        {   // synchronous transposing stage of the B tile
            const unsigned short* bp = B + (size_t)(k0 + bk) * N + bn + bnb;
#pragma unroll
            for (int i = 0; i < EPT; ++i) Bs[bnb + i][bk] = bp[i];
        }
        asm volatile("s_wait_asynccnt 0" ::: "memory");
        __syncthreads();

        BfFrag af;
        const int am  = wave_m * 16 + ln;
        const int bn0 = wave_n * WN + ln;
#pragma unroll
        for (int v = 0; v < 8; ++v) {
            int kb = (v < 4) ? (2 * v + 8 * half) : (16 + 2 * (v - 4) + 8 * half);
            af.u[v] = *(const unsigned int*)&As[am][kb];
        }
#pragma unroll
        for (int t = 0; t < NT; ++t) {
            BfFrag bf;
#pragma unroll
            for (int v = 0; v < 8; ++v) {
                int kb = (v < 4) ? (2 * v + 8 * half) : (16 + 2 * (v - 4) + 8 * half);
                bf.u[v] = *(const unsigned int*)&Bs[bn0 + t * 16][kb];
            }
            acc[t] = __builtin_amdgcn_wmma_f32_16x16x32_bf16(
                false, af.v, false, bf.v, (short)0, acc[t], false, false);
        }
    }

#pragma unroll
    for (int t = 0; t < NT; ++t) {
#pragma unroll
        for (int r = 0; r < 8; ++r) {
            int gm = bm + wave_m * 16 + half * 8 + r;
            int gn = bn + wave_n * WN + t * 16 + ln;
            float v0 = acc[t][r];
            if (EPILOGUE == 1) {
                v0 += bias[gn];
                v0 = (v0 > 20.0f) ? v0 : log1pf(__expf(v0));
            }
            C[(size_t)gm * N + gn] = v0;
        }
    }
}

// causal depthwise conv1d (K=4) + bias + SiLU; writes f32 (for the scan)
// and bf16 (GEMM operand) copies of xc.
__global__ void conv_silu_kernel(const float* __restrict__ xz,
                                 const float* __restrict__ w,
                                 const float* __restrict__ b,
                                 float* __restrict__ xc,
                                 unsigned short* __restrict__ xc_bf) {
    int idx = blockIdx.x * blockDim.x + threadIdx.x;
    if (idx >= BATCH * SEQ * INNER) return;
    int c  = idx & (INNER - 1);
    int l  = (idx >> 11) & (SEQ - 1);
    int bb = idx >> 22;
    float acc = b[c];
#pragma unroll
    for (int k = 0; k < KCONV; ++k) {
        int ls = l - (KCONV - 1) + k;
        if (ls >= 0)
            acc += w[c * KCONV + k] * xz[(size_t)(bb * SEQ + ls) * (2 * INNER) + c];
    }
    float s   = 1.0f / (1.0f + __expf(-acc));
    float val = acc * s;
    xc[idx]    = val;
    xc_bf[idx] = f2bf(val);
}

// selective scan: one thread per (batch, channel); 64-entry state in VGPRs,
// B/C vectors broadcast per timestep via LDS. Fuses a=exp(-dt), dtx, D-skip,
// silu(z) gate; emits bf16 directly (only the final GEMM consumes y).
__global__ __launch_bounds__(256)
void scan_kernel(const float* __restrict__ dt, const float* __restrict__ xc,
                 const float* __restrict__ Bm, const float* __restrict__ Cm,
                 const float* __restrict__ Dv, const float* __restrict__ xz,
                 unsigned short* __restrict__ y_bf) {
    __shared__ float sB[STATE];
    __shared__ float sC[STATE];
    const int tid = threadIdx.x;
    const int bpb = INNER / 256;
    const int b   = blockIdx.x / bpb;
    const int c   = (blockIdx.x % bpb) * 256 + tid;

    float s[STATE];
#pragma unroll
    for (int n = 0; n < STATE; ++n) s[n] = 0.0f;
    const float dcoef = Dv[c];

    for (int l = 0; l < SEQ; ++l) {
        const size_t row = (size_t)(b * SEQ + l);
        if (tid < STATE)          sB[tid]         = Bm[row * STATE + tid];
        else if (tid < 2 * STATE) sC[tid - STATE] = Cm[row * STATE + (tid - STATE)];
        __syncthreads();

        float dtv = dt[row * INNER + c];
        float xv  = xc[row * INNER + c];
        float a   = __expf(-dtv);
        float dx  = dtv * xv;
        float acc = 0.0f;
#pragma unroll
        for (int n = 0; n < STATE; ++n) {
            s[n] = fmaf(a, s[n], dx * sB[n]);
            acc  = fmaf(s[n], sC[n], acc);
        }
        float zv   = xz[row * (2 * INNER) + INNER + c];
        float gate = zv / (1.0f + __expf(-zv));
        y_bf[row * INNER + c] = f2bf((acc + dcoef * xv) * gate);
        __syncthreads();
    }
}

extern "C" void kernel_launch(void* const* d_in, const int* in_sizes, int n_in,
                              void* d_out, int out_size, void* d_ws, size_t ws_size,
                              hipStream_t stream) {
    const float* x      = (const float*)d_in[0];
    const float* W_in   = (const float*)d_in[1];
    const float* conv_w = (const float*)d_in[2];
    const float* conv_b = (const float*)d_in[3];
    const float* W_dt   = (const float*)d_in[4];
    const float* b_dt   = (const float*)d_in[5];
    const float* W_B    = (const float*)d_in[6];
    const float* W_C    = (const float*)d_in[7];
    const float* Dv     = (const float*)d_in[8];
    const float* W_out  = (const float*)d_in[9];
    float* out = (float*)d_out;

    const int M = BATCH * SEQ;                 // 4096
    const size_t E_XZ = (size_t)M * 2 * INNER; // 16.78M
    const size_t E_I  = (size_t)M * INNER;     // 8.39M
    const size_t E_S  = (size_t)M * STATE;     // 0.26M

    // f32 region
    float* xz = (float*)d_ws;
    float* xc = xz + E_XZ;
    float* dt = xc + E_I;
    float* Bm = dt + E_I;
    float* Cm = Bm + E_S;
    // bf16 region
    unsigned short* x_bf    = (unsigned short*)(Cm + E_S);
    unsigned short* Win_bf  = x_bf   + (size_t)HIDDEN * M;        // x is M*HIDDEN
    unsigned short* Wdt_bf  = Win_bf + (size_t)HIDDEN * 2 * INNER;
    unsigned short* WB_bf   = Wdt_bf + (size_t)INNER * INNER;
    unsigned short* WC_bf   = WB_bf  + (size_t)INNER * STATE;
    unsigned short* Wout_bf = WC_bf  + (size_t)INNER * STATE;
    unsigned short* xc_bf   = Wout_bf + (size_t)INNER * HIDDEN;
    unsigned short* y_bf    = xc_bf  + E_I;

    dim3 blk(256);
    auto cvt = [&](const float* src, unsigned short* dst, size_t n) {
        cvt_bf16_kernel<<<(unsigned)((n / 4 + 255) / 256), blk, 0, stream>>>(src, dst, (int)n);
    };

    // 0) one-time bf16 conversions (weights + input)
    cvt(x,     x_bf,    (size_t)M * HIDDEN);
    cvt(W_in,  Win_bf,  (size_t)HIDDEN * 2 * INNER);
    cvt(W_dt,  Wdt_bf,  (size_t)INNER * INNER);
    cvt(W_B,   WB_bf,   (size_t)INNER * STATE);
    cvt(W_C,   WC_bf,   (size_t)INNER * STATE);
    cvt(W_out, Wout_bf, (size_t)INNER * HIDDEN);

    // 1) xz = x @ W_in
    gemm_bf16_wmma<0, 64><<<dim3((2 * INNER) / 128, M / 64), blk, 0, stream>>>(
        x_bf, Win_bf, nullptr, xz, M, 2 * INNER, HIDDEN);

    // 2) xc = silu(causal_conv(x_inner) + conv_b)   (f32 + bf16 copies)
    int n1 = BATCH * SEQ * INNER;
    conv_silu_kernel<<<(n1 + 255) / 256, blk, 0, stream>>>(xz, conv_w, conv_b, xc, xc_bf);

    // 3) dt = softplus(xc @ W_dt + b_dt)
    gemm_bf16_wmma<1, 64><<<dim3(INNER / 128, M / 64), blk, 0, stream>>>(
        xc_bf, Wdt_bf, b_dt, dt, M, INNER, INNER);

    // 4) Bm = xc @ W_B ; Cm = xc @ W_C   (N=64 -> WN=32 variant)
    gemm_bf16_wmma<0, 32><<<dim3(1, M / 64), blk, 0, stream>>>(
        xc_bf, WB_bf, nullptr, Bm, M, STATE, INNER);
    gemm_bf16_wmma<0, 32><<<dim3(1, M / 64), blk, 0, stream>>>(
        xc_bf, WC_bf, nullptr, Cm, M, STATE, INNER);

    // 5) selective scan + skip + output gate -> y (bf16)
    scan_kernel<<<(BATCH * INNER) / 256, blk, 0, stream>>>(dt, xc, Bm, Cm, Dv, xz, y_bf);

    // 6) out = y @ W_out
    gemm_bf16_wmma<0, 64><<<dim3(HIDDEN / 128, M / 64), blk, 0, stream>>>(
        y_bf, Wout_bf, nullptr, out, M, HIDDEN, INNER);
}
